// SingleRenderer_43447889166611
// MI455X (gfx1250) — matrix operand order repaired and checked by reading, not compiled, not run on hardware
//
#include <hip/hip_runtime.h>
#include <math.h>

typedef _Float16 f16;
typedef __attribute__((ext_vector_type(16))) _Float16 v16h;
typedef __attribute__((ext_vector_type(8)))  _Float16 v8h;
typedef __attribute__((ext_vector_type(8)))  float    v8f;

#define RAYS   64
#define BLOCK  128
#define NWAVES 4
#define NS0    64
#define NSMAX  128
#define H1S    72   // halfs per staging row (64 + 8 pad)

// ---- dynamic LDS layout (bytes), all 16B aligned ----
#define OFF_D      0        // 64*128 f32
#define OFF_SDF    32768    // 64*128 f32
#define OFF_VW     65536    // 64*128 f32
#define OFF_ODIR   98304    // 64*8  f32 (o.xyz, dir.xyz, pad)
#define OFF_RGBA   100352   // 64*4  f32
#define OFF_NEWP   101376   // 64*16 u16
#define OFF_W1G    103424   // 3*64 f32
#define OFF_B1G    104192   // 64 f32
#define OFF_W3G    104448   // 64 f32
#define OFF_B3G    104704   // pad16
#define OFF_W1C    104720   // 6*64 f32
#define OFF_B1C    106256   // 64 f32
#define OFF_B2G    106512   // 64 f32
#define OFF_B2C    106768   // 64 f32
#define OFF_W3C    107024   // 64*3 f32
#define OFF_B3C    107792   // pad16
#define OFF_WT2G   107808   // 64*64 f16 (transposed [n][k])
#define OFF_WT2C   116000   // 64*64 f16 (transposed [n][k])
#define OFF_STAGE  124192   // 4 waves * 2 * 16*H1S halfs
#define SMEM_TOTAL 142624

__device__ __forceinline__ void lds_fence() {
  asm volatile("s_wait_dscnt 0x0" ::: "memory");
}
__device__ __forceinline__ float sigm(float x) { return 1.f / (1.f + __expf(-x)); }

// A fragment: 16x32 f16 activations from LDS row-major [16][H1S]
__device__ __forceinline__ v16h load_afrag(const f16* h1, int lane, int kc) {
  int m  = lane & 15;
  int kb = (lane >> 4) << 3;            // lanes 16-31 handle K-offset +8
  const f16* row = h1 + m * H1S + kc + kb;
  v8h lo = *(const v8h*)row;            // K = kc+kb+0..7
  v8h hi = *(const v8h*)(row + 16);     // K = kc+kb+16..23
  v16h r;
#pragma unroll
  for (int i = 0; i < 8; i++) { r[i] = lo[i]; r[i + 8] = hi[i]; }
  return r;
}
// B fragment: 32x16 f16 weights from LDS transposed [n][k] (stride 64)
__device__ __forceinline__ v16h load_bfrag(const f16* Wt, int lane, int nt, int kc) {
  int n  = (lane & 15) + (nt << 4);
  int ks = kc + ((lane >> 4) << 3);     // lanes 16-31: K +8
  const v8h* p = (const v8h*)(Wt + n * 64 + ks);
  v8h lo = p[0]; v8h hi = p[2];
  v16h r;
#pragma unroll
  for (int i = 0; i < 8; i++) { r[i] = lo[i]; r[i + 8] = hi[i]; }
  return r;
}

__global__ __launch_bounds__(BLOCK, 1) void neus_render_kernel(
    const float* __restrict__ rays_o, const float* __restrict__ rays_d,
    const float* __restrict__ Wg1, const float* __restrict__ bg1,
    const float* __restrict__ Wg2, const float* __restrict__ bg2,
    const float* __restrict__ Wg3, const float* __restrict__ bg3,
    const float* __restrict__ Wc1, const float* __restrict__ bc1,
    const float* __restrict__ Wc2, const float* __restrict__ bc2,
    const float* __restrict__ Wc3, const float* __restrict__ bc3,
    const float* __restrict__ ln_s,
    float* __restrict__ out, int nrays)
{
  extern __shared__ char sm[];
  float* dS   = (float*)(sm + OFF_D);
  float* sS   = (float*)(sm + OFF_SDF);
  float* vwS  = (float*)(sm + OFF_VW);
  float* odir = (float*)(sm + OFF_ODIR);
  float* rgba = (float*)(sm + OFF_RGBA);
  unsigned short* newp = (unsigned short*)(sm + OFF_NEWP);
  float* W1G = (float*)(sm + OFF_W1G);
  float* B1G = (float*)(sm + OFF_B1G);
  float* W3G = (float*)(sm + OFF_W3G);
  float* B3G = (float*)(sm + OFF_B3G);
  float* W1C = (float*)(sm + OFF_W1C);
  float* B1C = (float*)(sm + OFF_B1C);
  float* B2G = (float*)(sm + OFF_B2G);
  float* B2C = (float*)(sm + OFF_B2C);
  float* W3C = (float*)(sm + OFF_W3C);
  float* B3C = (float*)(sm + OFF_B3C);
  f16* WT2G = (f16*)(sm + OFF_WT2G);
  f16* WT2C = (f16*)(sm + OFF_WT2C);

  const int tid = threadIdx.x, lane = tid & 31, wave = tid >> 5;
  f16* h1 = (f16*)(sm + OFF_STAGE) + wave * (2 * 16 * H1S);
  f16* h2 = h1 + 16 * H1S;

  // ---- stage weights into LDS (layer-2 matrices transposed, f16) ----
  for (int i = tid; i < 3 * 64; i += BLOCK) W1G[i] = Wg1[i];
  for (int i = tid; i < 64; i += BLOCK) {
    B1G[i] = bg1[i]; B2G[i] = bg2[i]; W3G[i] = Wg3[i];
    B1C[i] = bc1[i]; B2C[i] = bc2[i];
  }
  for (int i = tid; i < 6 * 64; i += BLOCK) W1C[i] = Wc1[i];
  for (int i = tid; i < 192; i += BLOCK) W3C[i] = Wc3[i];
  if (tid == 0) { B3G[0] = bg3[0]; B3C[0] = bc3[0]; B3C[1] = bc3[1]; B3C[2] = bc3[2]; }
  for (int i = tid; i < 64 * 64; i += BLOCK) {
    int n = i >> 6, k = i & 63;
    WT2G[i] = (f16)Wg2[k * 64 + n];
    WT2C[i] = (f16)Wc2[k * 64 + n];
  }

  // ---- per-ray setup: normalize dir, near/far, coarse linspace samples ----
  if (tid < RAYS) {
    int gr = blockIdx.x * RAYS + tid;
    float ox = rays_o[gr * 3], oy = rays_o[gr * 3 + 1], oz = rays_o[gr * 3 + 2];
    float dx = rays_d[gr * 3], dy = rays_d[gr * 3 + 1], dz = rays_d[gr * 3 + 2];
    float inv = 1.f / sqrtf(dx * dx + dy * dy + dz * dz);
    dx *= inv; dy *= inv; dz *= inv;
    float midv  = -(ox * dx + oy * dy + oz * dz);
    float nearv = fmaxf(midv - 1.f, 0.001f);
    float farv  = midv + 1.f;
    float* op = odir + tid * 8;
    op[0] = ox; op[1] = oy; op[2] = oz; op[3] = dx; op[4] = dy; op[5] = dz;
    float* dr = dS + tid * NSMAX;
    for (int s2 = 0; s2 < NS0; s2++) {
      float t = (float)s2 * (1.f / 63.f);
      dr[s2] = nearv * (1.f - t) + farv * t;
    }
  }
  __syncthreads();

  // ---- cooperative MLP evaluation over 16-point tiles (WMMA f16, f32 acc) ----
  // mode 0: SDF at samples [0,64); mode 1: SDF at the 16 merged fine samples;
  // mode 2: RGB at 128 midpoints (last one padded, contribution skipped)
  auto eval = [&](int mode) {
    const float* W1 = (mode == 2) ? W1C : W1G;
    const float* B1 = (mode == 2) ? B1C : B1G;
    const float* B2 = (mode == 2) ? B2C : B2G;
    const f16*   WT = (mode == 2) ? WT2C : WT2G;
    int nT = (mode == 0) ? (RAYS * NS0 / 16) : (mode == 1) ? RAYS : (RAYS * NSMAX / 16);

    // hoist the 8 layer-2 weight fragments into registers for the whole phase
    v16h bw0[4], bw1[4];
#pragma unroll
    for (int nt4 = 0; nt4 < 4; nt4++) {
      bw0[nt4] = load_bfrag(WT, lane, nt4, 0);
      bw1[nt4] = load_bfrag(WT, lane, nt4, 32);
    }
    float bias2[4];
#pragma unroll
    for (int nt4 = 0; nt4 < 4; nt4++) bias2[nt4] = B2[(lane & 15) + (nt4 << 4)];

    for (int t = wave; t < nT; t += NWAVES) {
      // layer 1 (tiny K): each lane fills 2 columns per point, f16 into staging
      for (int pt = 0; pt < 16; pt++) {
        int ray, samp;
        if (mode == 0)      { ray = t >> 2; samp = ((t & 3) << 4) + pt; }
        else if (mode == 1) { ray = t;      samp = (int)newp[ray * 16 + pt]; }
        else                { ray = t >> 3; samp = ((t & 7) << 4) + pt; }
        const float* op = odir + ray * 8;
        float dd;
        if (mode == 2) {
          int i2 = samp < 127 ? samp : 126;
          dd = 0.5f * (dS[ray * NSMAX + i2] + dS[ray * NSMAX + i2 + 1]);
        } else dd = dS[ray * NSMAX + samp];
        float px = op[0] + dd * op[3], py = op[1] + dd * op[4], pz = op[2] + dd * op[5];
#pragma unroll
        for (int c2 = 0; c2 < 2; c2++) {
          int col = lane + (c2 << 5);
          float a = B1[col] + px * W1[col] + py * W1[64 + col] + pz * W1[128 + col];
          if (mode == 2)
            a += op[3] * W1[192 + col] + op[4] * W1[256 + col] + op[5] * W1[320 + col];
          h1[pt * H1S + col] = (f16)fmaxf(a, 0.f);
        }
      }
      lds_fence();
      // layer 2: 16x64 = 4 N-tiles x 2 K-chunk WMMAs, bias + relu on C fragments
      v16h a0 = load_afrag(h1, lane, 0);
      v16h a1 = load_afrag(h1, lane, 32);
#pragma unroll
      for (int nt4 = 0; nt4 < 4; nt4++) {
        int n = (lane & 15) + (nt4 << 4);
        float bv = bias2[nt4];
        v8f c;
#pragma unroll
        for (int j = 0; j < 8; j++) c[j] = bv;
        c = __builtin_amdgcn_wmma_f32_16x16x32_f16(false, a0, false, bw0[nt4], (short)0, c, false, false);
        c = __builtin_amdgcn_wmma_f32_16x16x32_f16(false, a1, false, bw1[nt4], (short)0, c, false, false);
        int mb = (lane >> 4) << 3;
#pragma unroll
        for (int j = 0; j < 8; j++) h2[(mb + j) * H1S + n] = (f16)fmaxf(c[j], 0.f);
      }
      lds_fence();
      // layer 3 heads: 2 lanes per point, half-wave K split + shfl_xor(16) reduce
      int pt = lane & 15;
      int kh = (lane >> 4) << 5;
      int ray, samp;
      if (mode == 0)      { ray = t >> 2; samp = ((t & 3) << 4) + pt; }
      else if (mode == 1) { ray = t;      samp = (int)newp[ray * 16 + pt]; }
      else                { ray = t >> 3; samp = ((t & 7) << 4) + pt; }
      const f16* hr = h2 + pt * H1S + kh;
      if (mode != 2) {
        float s = 0.f;
#pragma unroll
        for (int q = 0; q < 32; q++) s += (float)hr[q] * W3G[kh + q];
        s += __shfl_xor(s, 16, 32);
        if (lane < 16) {
          const float* op = odir + ray * 8;
          float dd = dS[ray * NSMAX + samp];
          float px = op[0] + dd * op[3], py = op[1] + dd * op[4], pz = op[2] + dd * op[5];
          sS[ray * NSMAX + samp] =
              0.7f - sqrtf(px * px + py * py + pz * pz) + 0.1f * (s + B3G[0]);
        }
      } else {
        float s0 = 0.f, s1 = 0.f, s2 = 0.f;
#pragma unroll
        for (int q = 0; q < 32; q++) {
          float h = (float)hr[q]; int k = kh + q;
          s0 += h * W3C[k * 3]; s1 += h * W3C[k * 3 + 1]; s2 += h * W3C[k * 3 + 2];
        }
        s0 += __shfl_xor(s0, 16, 32);
        s1 += __shfl_xor(s1, 16, 32);
        s2 += __shfl_xor(s2, 16, 32);
        if (lane < 16 && samp < 127) {
          float wv = vwS[ray * NSMAX + samp];
          atomicAdd(&rgba[ray * 4 + 0], wv * sigm(s0 + B3C[0]));
          atomicAdd(&rgba[ray * 4 + 1], wv * sigm(s1 + B3C[1]));
          atomicAdd(&rgba[ray * 4 + 2], wv * sigm(s2 + B3C[2]));
        }
      }
    }
  };

  eval(0);
  __syncthreads();

  // ---- 4 up-sampling rounds: one ray per lane ----
  for (int it = 0; it < 4; it++) {
    if (tid < RAYS) {
      int n = NS0 + 16 * it;
      float* dr = dS + tid * NSMAX;
      float* sr = sS + tid * NSMAX;
      float w[127], cdf[128], fine[16];
      float T = 1.f, pdot = 0.f, wsum = 0.f;
      float sphi = 256.f * (float)(1 << it);
      for (int k = 0; k < n - 1; k++) {
        float ps = sr[k], ns = sr[k + 1], pz = dr[k], nz = dr[k + 1];
        float midv = 0.5f * (ps + ns);
        float dt = (ns - ps) / (nz - pz + 1e-5f);
        float dc = fminf(pdot, dt); pdot = dt;           // prev_dot uses unclipped value
        dc = fminf(fmaxf(dc, -10.f), 0.f);
        float dist = nz - pz;
        float pe = midv - dc * dist * 0.5f, ne = midv + dc * dist * 0.5f;
        float pc = sigm(pe * sphi), nc = sigm(ne * sphi);
        float al = (pc - nc + 1e-5f) / (pc + 1e-5f);
        float wk = al * T + 1e-5f;                       // +1e-5 from sample_pdf
        T *= (1.f - al + 1e-10f);
        w[k] = wk; wsum += wk;
      }
      float invs = 1.f / wsum, acc = 0.f;
      cdf[0] = 0.f;
      for (int k = 0; k < n - 1; k++) { acc += w[k] * invs; cdf[k + 1] = acc; }
      int idx = 0;
      for (int j = 0; j < 16; j++) {
        float u = (float)j * (1.f / 15.f);
        while (idx < n && cdf[idx] <= u) idx++;          // searchsorted right
        int below = idx - 1; below = below < 0 ? 0 : (below > n - 1 ? n - 1 : below);
        int above = idx > n - 1 ? n - 1 : idx;
        float cb = cdf[below], ca = cdf[above];
        float den = ca - cb; if (den < 1e-5f) den = 1.f;
        float bb = dr[below], ba = dr[above];
        fine[j] = bb + (u - cb) / den * (ba - bb);
      }
      // stable backward in-place merge of sorted fine samples; record positions
      int ii = n - 1, jj = 15;
      for (int k = n + 15; k >= 0; k--) {
        bool tn = (ii < 0) || (jj >= 0 && fine[jj] >= dr[ii]);
        if (tn) { dr[k] = fine[jj]; sr[k] = 0.f; newp[tid * 16 + jj] = (unsigned short)k; jj--; }
        else    { dr[k] = dr[ii];   sr[k] = sr[ii]; ii--; }
      }
    }
    __syncthreads();
    eval(1);
    __syncthreads();
  }

  // ---- final compositing weights, depth and acc outputs ----
  float smodel = __expf(ln_s[0] * 10.f);
  if (tid < RAYS) {
    int gr = blockIdx.x * RAYS + tid;
    float* dr = dS + tid * NSMAX;
    float* sr = sS + tid * NSMAX;
    float* vr = vwS + tid * NSMAX;
    float c0 = sigm(sr[0] * smodel);
    float T = 1.f, wsum = 0.f, dsum = 0.f;
    for (int k = 0; k < 127; k++) {
      float c1 = sigm(sr[k + 1] * smodel);
      float a = fmaxf((c0 - c1) / (c0 + 1e-10f), 0.f);
      float wv = a * T; T *= (1.f - a + 1e-10f);
      vr[k] = wv; wsum += wv; dsum += wv * 0.5f * (dr[k] + dr[k + 1]);
      c0 = c1;
    }
    out[nrays * 3 + gr] = dsum / (wsum + 1e-10f);
    out[nrays * 4 + gr] = wsum;
    rgba[tid * 4 + 0] = 0.f; rgba[tid * 4 + 1] = 0.f; rgba[tid * 4 + 2] = 0.f;
  }
  __syncthreads();

  eval(2);
  __syncthreads();

  if (tid < RAYS) {
    int gr = blockIdx.x * RAYS + tid;
    out[gr * 3 + 0] = rgba[tid * 4 + 0];
    out[gr * 3 + 1] = rgba[tid * 4 + 1];
    out[gr * 3 + 2] = rgba[tid * 4 + 2];
  }
}

extern "C" void kernel_launch(void* const* d_in, const int* in_sizes, int n_in,
                              void* d_out, int out_size, void* d_ws, size_t ws_size,
                              hipStream_t stream) {
  (void)n_in; (void)out_size; (void)d_ws; (void)ws_size;
  const float* rays_o = (const float*)d_in[0];
  const float* rays_d = (const float*)d_in[1];
  const float* Wg1 = (const float*)d_in[2];
  const float* bg1 = (const float*)d_in[3];
  const float* Wg2 = (const float*)d_in[4];
  const float* bg2 = (const float*)d_in[5];
  const float* Wg3 = (const float*)d_in[6];
  const float* bg3 = (const float*)d_in[7];
  const float* Wc1 = (const float*)d_in[8];
  const float* bc1 = (const float*)d_in[9];
  const float* Wc2 = (const float*)d_in[10];
  const float* bc2 = (const float*)d_in[11];
  const float* Wc3 = (const float*)d_in[12];
  const float* bc3 = (const float*)d_in[13];
  const float* lns = (const float*)d_in[14];
  float* out = (float*)d_out;
  int nrays = in_sizes[0] / 3;
  int blocks = nrays / RAYS;
  hipLaunchKernelGGL(neus_render_kernel, dim3(blocks), dim3(BLOCK), SMEM_TOTAL, stream,
                     rays_o, rays_d, Wg1, bg1, Wg2, bg2, Wg3, bg3,
                     Wc1, bc1, Wc2, bc2, Wc3, bc3, lns, out, nrays);
}
